// GenPhiloTextB_62869731279074
// MI455X (gfx1250) — matrix operand order repaired and checked by reading, not compiled, not run on hardware
//
#include <hip/hip_runtime.h>
#include <hip/hip_bf16.h>
#include <cstdint>
#include <cstddef>

// Problem constants (match reference)
#define BB 256   // batch
#define TT 256   // timesteps
#define EE 64    // embedding dim
#define NN 512   // hidden units
#define GG 2048  // 4*N gate dim
#define VV 128   // vocab
#define KT_Z 18  // (EE+NN)/32 K-tiles for fused input+recurrent GEMM
#define CT_Z 128 // GG/16 column tiles
#define KT_D 16  // NN/32 K-tiles for output GEMM
#define CT_D 8   // VV/16 column tiles

typedef __attribute__((ext_vector_type(16))) __bf16 v16bf;
typedef __attribute__((ext_vector_type(8)))  float  v8f;
typedef __attribute__((ext_vector_type(4)))  unsigned int v4u;
typedef __attribute__((ext_vector_type(8)))  unsigned int v8u;

union V16 { uint4 q[2]; v16bf v; };

__device__ __forceinline__ unsigned short f2bf(float f) {
    union { float f; unsigned int u; } a; a.f = f;
    unsigned int u = a.u;
    unsigned int r = (u + 0x7FFFu + ((u >> 16) & 1u)) >> 16; // RNE
    return (unsigned short)r;
}

// Branch-free activations on v_exp_f32 / v_rcp_f32 (no libm exec-mask paths)
__device__ __forceinline__ float sigmoid_fast(float x) {
    return __builtin_amdgcn_rcpf(1.0f + __expf(-x));
}
__device__ __forceinline__ float tanh_fast(float x) {
    // 1 - 2/(1+e^{2x}) : saturates to +/-1, no branches
    return fmaf(-2.0f, __builtin_amdgcn_rcpf(1.0f + __expf(2.0f * x)), 1.0f);
}

// ---- Tensor Data Mover: 1-D contiguous copy global -> LDS (8-byte elements) ----
// D# per CDNA5 ISA. The LDS pointer is passed as an
// (unreferenced) operand so the shared array's address escapes into the asm:
// with the "memory" clobber the compiler must assume the TDM writes it, which
// keeps the subsequent ds_load_b128 reads alive.
__device__ __forceinline__ void tdm_load_1d(void* lds_ptr,
                                            unsigned lds_byte_off,
                                            const void* gptr,
                                            unsigned nelem8 /* #8-byte elements */) {
    unsigned long long ga = (unsigned long long)(uintptr_t)gptr;
    v4u g0;
    g0[0] = 0x1u;                                             // count=1, user mode
    g0[1] = lds_byte_off;                                     // lds_addr
    g0[2] = (unsigned)(ga & 0xFFFFFFFFu);                     // global_addr[31:0]
    g0[3] = (unsigned)((ga >> 32) & 0x1FFFFFFu) | (2u << 30); // addr[56:32] | type=2
    v8u g1;
    g1[0] = (3u << 16);                       // workgroup_mask=0, data_size=3 (8B)
    g1[1] = (nelem8 & 0xFFFFu) << 16;         // tensor_dim0[15:0]
    g1[2] = ((nelem8 >> 16) & 0xFFFFu) | (1u << 16); // tensor_dim0 hi, tensor_dim1=1
    g1[3] = (nelem8 & 0xFFFFu) << 16;         // tile_dim0 = nelem8
    g1[4] = 1u;                               // tile_dim1=1, tile_dim2=0
    g1[5] = nelem8;                           // tensor_dim0_stride[31:0]
    g1[6] = (nelem8 & 0xFFFFu) << 16;         // stride0 hi=0, tensor_dim1_stride lo
    g1[7] = (nelem8 >> 16);                   // tensor_dim1_stride[47:16]
    asm volatile("tensor_load_to_lds %0, %1"
                 :
                 : "s"(g0), "s"(g1), "s"(lds_ptr)
                 : "memory");
}

// ---------------- pack kernels ----------------

// Wz = concat_K(Wx[64,2048], Wh[512,2048]) -> bf16, WMMA-B tile layout
// layout: [ct(128)][kt(18)][lane(32)][j(16)] halves, tile = 32x16 (KxN)
__global__ __launch_bounds__(256) void pack_wz(const float* __restrict__ Wx,
                                               const float* __restrict__ Wh,
                                               unsigned short* __restrict__ Wzp) {
    int tid = blockIdx.x * 256 + threadIdx.x;           // 128*18*512 = 1,179,648
    if (tid >= CT_Z * KT_Z * 512) return;
    int j      = tid & 15;
    int lane   = (tid >> 4) & 31;
    int kt     = (tid >> 9) % KT_Z;
    int ct     = tid / (KT_Z * 512);
    int laneHi = lane >> 4, laneLo = lane & 15;
    int k_in   = laneHi * 8 + (j & 7) + ((j >> 3) << 4); // per-lane K pattern
    int k      = kt * 32 + k_in;
    int g      = ct * 16 + laneLo;
    float v    = (k < EE) ? Wx[(size_t)k * GG + g] : Wh[(size_t)(k - EE) * GG + g];
    Wzp[tid] = f2bf(v);
}

// Wd[512,128] -> bf16 WMMA-B tiles: [ct(8)][kt(16)][lane(32)][j(16)]
__global__ __launch_bounds__(256) void pack_wd(const float* __restrict__ Wd,
                                               unsigned short* __restrict__ Wdp) {
    int tid = blockIdx.x * 256 + threadIdx.x;           // 8*16*512 = 65536
    if (tid >= CT_D * KT_D * 512) return;
    int j      = tid & 15;
    int lane   = (tid >> 4) & 31;
    int kt     = (tid >> 9) % KT_D;
    int ct     = tid / (KT_D * 512);
    int laneHi = lane >> 4, laneLo = lane & 15;
    int k_in   = laneHi * 8 + (j & 7) + ((j >> 3) << 4);
    int k      = kt * 32 + k_in;
    int v      = ct * 16 + laneLo;
    Wdp[tid] = f2bf(Wd[(size_t)k * VV + v]);
}

__global__ __launch_bounds__(256) void pack_emb(const float* __restrict__ emb,
                                                unsigned short* __restrict__ embb) {
    int tid = blockIdx.x * 256 + threadIdx.x;           // 128*64 = 8192
    if (tid < VV * EE) embb[tid] = f2bf(emb[tid]);
}

// hs slot 0 = bf16(h_0); c_ws = c_0
__global__ __launch_bounds__(256) void init_state(const float* __restrict__ h0,
                                                  const float* __restrict__ c0,
                                                  unsigned short* __restrict__ hs,
                                                  float* __restrict__ c_ws) {
    int tid = blockIdx.x * 256 + threadIdx.x;           // 256*512 = 131072
    if (tid < BB * NN) {
        c_ws[tid] = c0[tid];
        hs[tid]   = f2bf(h0[tid]);
    }
}

// 4 gate-tile WMMAs for one K-step, B fragments from LDS
__device__ __forceinline__ void gate_wmma4(const unsigned short* bp,
                                           const v16bf& av, v8f* acc) {
    #pragma unroll
    for (int g = 0; g < 4; ++g) {
        V16 b;
        b.q[0] = *(const uint4*)(bp + (size_t)g * (KT_Z * 512));     // ds_load_b128
        b.q[1] = *(const uint4*)(bp + (size_t)g * (KT_Z * 512) + 8);
        acc[g] = __builtin_amdgcn_wmma_f32_16x16x32_bf16(
            false, av, false, b.v, (short)0, acc[g], false, false);
    }
}

// ---------------- per-timestep LSTM cell ----------------
// grid: 64 blocks x 256 threads. Block owns one unit-column tile cb (16 units,
// all 4 gates) and 8 batch-row tiles (one per wave). Wave 0 DMAs the block's
// four packed-B gate strips (4 x 18KB) into LDS with the Tensor Data Mover,
// then every wave runs K=576 bf16 WMMAs from LDS-resident B fragments.
__global__ __launch_bounds__(256) void lstm_step(const int* __restrict__ X,
                                                 const unsigned short* __restrict__ embb,
                                                 const unsigned short* __restrict__ Wzp,
                                                 const float* __restrict__ bias,
                                                 unsigned short* __restrict__ hs,
                                                 float* __restrict__ c_ws,
                                                 int t) {
    __shared__ __align__(16) unsigned short Bsh[4 * KT_Z * 512]; // 72 KB

    const int wave   = threadIdx.x >> 5;
    const int lane   = threadIdx.x & 31;
    const int cb     = blockIdx.x & 31;         // unit col tile [0,32)
    const int rtHalf = blockIdx.x >> 5;         // [0,2)
    const int rt     = rtHalf * 8 + wave;       // batch row tile [0,16)
    const int laneHi = lane >> 4;
    const int laneLo = lane & 15;

    // Stage B: 4 gate strips, each KT_Z*1KB contiguous in Wzp, via TDM.
    if (wave == 0) {
        #pragma unroll
        for (int g = 0; g < 4; ++g) {
            const int ct = cb + g * 32;
            tdm_load_1d((void*)Bsh,
                        (unsigned)(g * KT_Z * 1024),
                        Wzp + (size_t)ct * KT_Z * 512,
                        (KT_Z * 1024) / 8);     // 2304 8-byte elements
        }
        __builtin_amdgcn_s_wait_tensorcnt(0);
    }
    __syncthreads();

    const int rowBase = rt * 16;
    const int arow    = rowBase + laneLo;       // A-matrix row handled by this lane
    const int embIdx  = X[arow * TT + t];

    const unsigned short* aptr_x = embb + (size_t)embIdx * EE;
    const unsigned short* hprev  = hs + (size_t)t * (BB * NN);
    const unsigned short* aptr_h = hprev + (size_t)arow * NN;
    const unsigned short* bbase  = Bsh + lane * 16;

    v8f acc[4] = {v8f{0,0,0,0,0,0,0,0}, v8f{0,0,0,0,0,0,0,0},
                  v8f{0,0,0,0,0,0,0,0}, v8f{0,0,0,0,0,0,0,0}};

    // ---- peeled prologue: two embedding K-tiles, pipelined ----
    V16 cur, nxt;
    cur.q[0] = *(const uint4*)(aptr_x + laneHi * 8);
    cur.q[1] = *(const uint4*)(aptr_x + 16 + laneHi * 8);
    nxt.q[0] = *(const uint4*)(aptr_x + 32 + laneHi * 8);
    nxt.q[1] = *(const uint4*)(aptr_x + 48 + laneHi * 8);

    gate_wmma4(bbase, cur.v, acc);              // kt = 0
    cur = nxt;
    nxt.q[0] = *(const uint4*)(aptr_h + laneHi * 8);        // h chunk 0
    nxt.q[1] = *(const uint4*)(aptr_h + 16 + laneHi * 8);
    gate_wmma4(bbase + 512, cur.v, acc);        // kt = 1
    cur = nxt;

    // ---- steady state over the 16 h K-tiles: unconditional preload ----
    // (last preload over-reads one 64B chunk into the next hs slot: in-bounds)
    #pragma unroll 2
    for (int kt = 2; kt < KT_Z; ++kt) {
        const unsigned short* ap = aptr_h + (kt - 1) * 32;  // h chunk (kt-1)... next
        V16 pre;
        pre.q[0] = *(const uint4*)(ap + laneHi * 8);
        pre.q[1] = *(const uint4*)(ap + 16 + laneHi * 8);
        gate_wmma4(bbase + kt * 512, cur.v, acc);
        cur = pre;
    }

    const int n  = cb * 16 + laneLo;            // unit index (D-tile column)
    const float bi = bias[n];
    const float bf = bias[NN + n];
    const float bg = bias[2 * NN + n];
    const float bo = bias[3 * NN + n];

    unsigned short* hout = hs + (size_t)(t + 1) * (BB * NN);

    #pragma unroll
    for (int v = 0; v < 8; ++v) {
        const int m = rowBase + v + 8 * laneHi; // D-tile row -> batch index
        const float gi = sigmoid_fast(acc[0][v] + bi);
        const float gf = sigmoid_fast(acc[1][v] + bf);
        const float gg = tanh_fast(acc[2][v] + bg);
        const float go = sigmoid_fast(acc[3][v] + bo);
        const size_t idx = (size_t)m * NN + n;
        const float cn = gf * c_ws[idx] + gi * gg;
        c_ws[idx] = cn;
        hout[idx] = f2bf(go * tanh_fast(cn));
    }
}

// ---------------- output projection: logits = hs @ Wd + bd ----------------
// rows r = t*B + b over hs[T,B,N]; 4096 row-tiles x 8 col-tiles, 1 wave per tile
__global__ __launch_bounds__(256) void out_gemm(const unsigned short* __restrict__ hs,
                                                const unsigned short* __restrict__ Wdp,
                                                const float* __restrict__ bd,
                                                float* __restrict__ out) {
    const int wave   = threadIdx.x >> 5;
    const int lane   = threadIdx.x & 31;
    const int task   = blockIdx.x * 8 + wave;   // [0, 32768)
    const int rt     = task >> 3;               // row tile [0,4096)
    const int ct     = task & 7;                // col tile [0,8)
    const int laneHi = lane >> 4;
    const int laneLo = lane & 15;

    const int r  = rt * 16 + laneLo;            // global row for A loads
    const int t  = r >> 8;                      // B == 256
    const int b  = r & 255;
    const unsigned short* aptr = hs + ((size_t)(t + 1) * BB + b) * NN;
    const unsigned short* bptr = Wdp + (size_t)ct * KT_D * 512 + lane * 16;

    // 1-deep software pipeline on both operands; unconditional preload
    // (kt==15 preload over-reads one chunk, still inside the workspace carve)
    V16 a, bm;
    a.q[0]  = *(const uint4*)(aptr + laneHi * 8);
    a.q[1]  = *(const uint4*)(aptr + 16 + laneHi * 8);
    bm.q[0] = *(const uint4*)(bptr);
    bm.q[1] = *(const uint4*)(bptr + 8);

    v8f acc = v8f{0,0,0,0,0,0,0,0};
    #pragma unroll 2
    for (int kt = 0; kt < KT_D; ++kt) {
        const unsigned short* ap = aptr + (kt + 1) * 32;
        const unsigned short* bp = bptr + (kt + 1) * 512;
        V16 an, bn;
        an.q[0] = *(const uint4*)(ap + laneHi * 8);
        an.q[1] = *(const uint4*)(ap + 16 + laneHi * 8);
        bn.q[0] = *(const uint4*)(bp);
        bn.q[1] = *(const uint4*)(bp + 8);
        acc = __builtin_amdgcn_wmma_f32_16x16x32_bf16(
            false, a.v, false, bm.v, (short)0, acc, false, false);
        a = an; bm = bn;
    }

    const int col = ct * 16 + laneLo;
    const float bdv = bd[col];
    #pragma unroll
    for (int v = 0; v < 8; ++v) {
        const int rr = rt * 16 + v + 8 * laneHi;
        const int tt = rr >> 8;
        const int bb2 = rr & 255;
        out[((size_t)bb2 * TT + tt) * VV + col] = acc[v] + bdv;
    }
}

// ---------------- host launch ----------------

extern "C" void kernel_launch(void* const* d_in, const int* in_sizes, int n_in,
                              void* d_out, int out_size, void* d_ws, size_t ws_size,
                              hipStream_t stream) {
    (void)in_sizes; (void)n_in; (void)out_size; (void)ws_size;
    const int*   X   = (const int*)  d_in[0];
    const float* h0  = (const float*)d_in[1];
    const float* c0  = (const float*)d_in[2];
    const float* emb = (const float*)d_in[3];
    const float* Wx  = (const float*)d_in[4];
    const float* Wh  = (const float*)d_in[5];
    const float* bia = (const float*)d_in[6];
    const float* Wd  = (const float*)d_in[7];
    const float* bd  = (const float*)d_in[8];
    float* out = (float*)d_out;

    // workspace carve-up (256B aligned)
    char* ws = (char*)d_ws;
    size_t off = 0;
    auto carve = [&](size_t bytes) {
        void* p = ws + off;
        off += (bytes + 255) & ~(size_t)255;
        return p;
    };
    unsigned short* hs   = (unsigned short*)carve((size_t)(TT + 1) * BB * NN * 2); // 64.25 MB
    float*          c_ws = (float*)         carve((size_t)BB * NN * 4);            // 0.5 MB
    unsigned short* Wzp  = (unsigned short*)carve((size_t)CT_Z * KT_Z * 512 * 2);  // 2.25 MB
    unsigned short* Wdp  = (unsigned short*)carve((size_t)CT_D * KT_D * 512 * 2);  // 128 KB
    unsigned short* embb = (unsigned short*)carve((size_t)VV * EE * 2);            // 16 KB

    // one-time (per call) packing + state init
    pack_wz <<<(CT_Z * KT_Z * 512 + 255) / 256, 256, 0, stream>>>(Wx, Wh, Wzp);
    pack_wd <<<(CT_D * KT_D * 512 + 255) / 256, 256, 0, stream>>>(Wd, Wdp);
    pack_emb<<<(VV * EE + 255) / 256,            256, 0, stream>>>(emb, embb);
    init_state<<<(BB * NN + 255) / 256,          256, 0, stream>>>(h0, c0, hs, c_ws);

    // sequential recurrence: one launch per timestep (graph-captured)
    for (int t = 0; t < TT; ++t) {
        lstm_step<<<64, 256, 0, stream>>>(X, embb, Wzp, bia, hs, c_ws, t);
    }

    // final dense projection over all (t, b) rows
    out_gemm<<<4096, 256, 0, stream>>>(hs, Wdp, bd, out);
}
